// MoDBlock_89318139888016
// MI455X (gfx1250) — matrix operand order repaired and checked by reading, not compiled, not run on hardware
//
#include <hip/hip_runtime.h>
#include <hip/hip_bf16.h>
#include <math.h>

// ---------------- problem constants (match reference) ----------------
constexpr int Bc  = 8;
constexpr int Tc  = 2048;
constexpr int Cc  = 1024;
constexpr int Kc  = 1024;      // top_k
constexpr int NH  = 16;
constexpr int DHc = 64;
constexpr int DFFc = 4096;
constexpr long BK = (long)Bc * Kc;   // 8192 rows in reduced sequence

// ---------------- vector types ----------------
typedef __attribute__((ext_vector_type(16))) _Float16 v16h;
typedef __attribute__((ext_vector_type(8)))  _Float16 v8h;
typedef __attribute__((ext_vector_type(4)))  _Float16 v4h;
typedef __attribute__((ext_vector_type(8)))  float    v8f;
typedef __attribute__((ext_vector_type(4)))  float    v4f;

__device__ __forceinline__ v16h cat8(v8h lo, v8h hi) {
  return __builtin_shufflevector(lo, hi, 0,1,2,3,4,5,6,7,8,9,10,11,12,13,14,15);
}

// Fragment load from a per-lane base pointer: lo 8 halfs at p+kk, hi 8 halfs at p+kk+hiOff.
// A-fragment (16x32): hiOff=8 (16 contiguous K per lane-half, same k order as B)
// B-fragment (32x16, TN): hiOff=8 (16 contiguous K per lane-half)
__device__ __forceinline__ v16h frag_at(const _Float16* p, int kk, int hiOff) {
  v8h lo = *(const v8h*)(p + kk);
  v8h hi = *(const v8h*)(p + kk + hiOff);
  return cat8(lo, hi);
}

__device__ __forceinline__ v8f wmma16(v16h a, v16h b, v8f c) {
  return __builtin_amdgcn_wmma_f32_16x16x32_f16(false, a, false, b, (short)0, c, false, false);
}

// ---------------- 1) router logits: one wave per token, float4 loads ----------------
__global__ void __launch_bounds__(256) router_logits_kernel(
    const float* __restrict__ x, const float* __restrict__ wr, float* __restrict__ logits) {
  int wave = threadIdx.x >> 5, lane = threadIdx.x & 31;
  int wid = blockIdx.x * 8 + wave;          // global token id in [0, B*T)
  const float* row = x + (size_t)wid * Cc;
  float s = 0.f;
  #pragma unroll
  for (int i = lane * 4; i < Cc; i += 128) {
    v4f xv = *(const v4f*)(row + i);
    v4f wv = *(const v4f*)(wr + i);
    s += xv[0]*wv[0] + xv[1]*wv[1] + xv[2]*wv[2] + xv[3]*wv[3];
  }
  #pragma unroll
  for (int o = 16; o; o >>= 1) s += __shfl_xor(s, o, 32);
  if (lane == 0) logits[wid] = s;
}

// ---------------- 2) per-batch top-k via bitonic sort ----------------
__global__ void __launch_bounds__(1024) topk_kernel(
    const float* __restrict__ logits, int* __restrict__ idxOut, float* __restrict__ wOut) {
  __shared__ float v[Tc];
  __shared__ int   id[Tc];
  int b = blockIdx.x, tid = threadIdx.x;
  for (int i = tid; i < Tc; i += 1024) { v[i] = logits[b * Tc + i]; id[i] = i; }
  __syncthreads();
  // sort all 2048 by (value desc, index asc)
  for (int ksz = 2; ksz <= Tc; ksz <<= 1) {
    for (int j = ksz >> 1; j > 0; j >>= 1) {
      for (int i = tid; i < Tc; i += 1024) {
        int ixj = i ^ j;
        if (ixj > i) {
          float va = v[i], vb = v[ixj]; int ia = id[i], ib = id[ixj];
          bool before = (va > vb) || (va == vb && ia < ib);   // desired order
          bool up = ((i & ksz) == 0);
          if (up ? !before : before) { v[i] = vb; v[ixj] = va; id[i] = ib; id[ixj] = ia; }
        }
      }
      __syncthreads();
    }
  }
  // resort first K=1024 by index ascending
  for (int ksz = 2; ksz <= Kc; ksz <<= 1) {
    for (int j = ksz >> 1; j > 0; j >>= 1) {
      int i = tid, ixj = i ^ j;
      if (i < Kc && ixj > i && ixj < Kc) {
        bool before = id[i] < id[ixj];
        bool up = ((i & ksz) == 0);
        if (up ? !before : before) {
          float tv = v[i]; v[i] = v[ixj]; v[ixj] = tv;
          int ti = id[i]; id[i] = id[ixj]; id[ixj] = ti;
        }
      }
      __syncthreads();
    }
  }
  if (tid < Kc) { idxOut[b * Kc + tid] = id[tid]; wOut[b * Kc + tid] = v[tid]; }
}

// ---------------- block reduction helper (256 threads) ----------------
__device__ __forceinline__ float block_sum256(float s, float* red, int tid) {
  #pragma unroll
  for (int o = 16; o; o >>= 1) s += __shfl_xor(s, o, 32);
  if ((tid & 31) == 0) red[tid >> 5] = s;
  __syncthreads();
  float tot = 0.f;
  #pragma unroll
  for (int i = 0; i < 8; ++i) tot += red[i];
  __syncthreads();
  return tot;
}

// ---------------- 3) gather + LN1 (block per selected token), float4 ----------------
__global__ void __launch_bounds__(256) gather_ln1_kernel(
    const float* __restrict__ x, const int* __restrict__ idx,
    const float* __restrict__ g, const float* __restrict__ be,
    float* __restrict__ h, _Float16* __restrict__ aln) {
  __shared__ float red[8];
  int b = blockIdx.y, kk = blockIdx.x, tid = threadIdx.x;
  int tok = idx[b * Kc + kk];
  const float* src = x + ((size_t)b * Tc + tok) * Cc;
  size_t row = (size_t)(b * Kc + kk);
  float* hr = h + row * Cc;
  int c0 = tid * 4;
  v4f v = *(const v4f*)(src + c0);
  *(v4f*)(hr + c0) = v;
  float s = v[0] + v[1] + v[2] + v[3];
  float mean = block_sum256(s, red, tid) * (1.0f / Cc);
  float d = 0.f;
  #pragma unroll
  for (int i = 0; i < 4; ++i) { float t = v[i] - mean; d += t * t; }
  float var = block_sum256(d, red, tid) * (1.0f / Cc);
  float rstd = rsqrtf(var + 1e-5f);
  v4f gv = *(const v4f*)(g + c0);
  v4f bv = *(const v4f*)(be + c0);
  v4h o;
  #pragma unroll
  for (int i = 0; i < 4; ++i) o[i] = (_Float16)((v[i] - mean) * rstd * gv[i] + bv[i]);
  *(v4h*)(aln + row * Cc + c0) = o;
}

// ---------------- 4) f32 -> f16 convert (x4 vectorized) ----------------
__global__ void __launch_bounds__(256) f32_to_f16_kernel(
    const float* __restrict__ src, _Float16* __restrict__ dst, long n4) {
  long i = (long)blockIdx.x * 256 + threadIdx.x;
  long stride = (long)gridDim.x * 256;
  for (; i < n4; i += stride) {
    v4f v = *(const v4f*)(src + i * 4);
    v4h o;
    #pragma unroll
    for (int j = 0; j < 4; ++j) o[j] = (_Float16)v[j];
    *(v4h*)(dst + i * 4) = o;
  }
}

// ---------------- 5) WMMA TN GEMM, ping-pong software pipeline ----------------
// A: f16 [M,K] lda;  B: f16 [N,K] ldb;  D: f32 [M,N] ldd.  Block tile 128x64, wave tile 16x64.
__global__ void __launch_bounds__(256) wmma_gemm_tn(
    float* __restrict__ D, const _Float16* __restrict__ A, const _Float16* __restrict__ Bm,
    const float* __restrict__ bias, int M, int N, int K, int lda, int ldb, int ldd) {
  int wave = threadIdx.x >> 5, lane = threadIdx.x & 31;
  int m0 = blockIdx.y * 128 + wave * 16;
  int n0 = blockIdx.x * 64;
  if (m0 >= M || n0 >= N) return;

  // per-lane base pointers (inner loop: base + kk with immediate/induction offsets)
  const _Float16* pa = A + (size_t)(m0 + (lane & 15)) * lda + ((lane >> 4) << 4);
  const _Float16* pb[4];
  #pragma unroll
  for (int t = 0; t < 4; ++t)
    pb[t] = Bm + (size_t)(n0 + 16 * t + (lane & 15)) * ldb + ((lane >> 4) << 4);

  v8f acc[4] = {};
  // prologue: stage0 = k-step 0, stage1 = k-step 32
  v16h a0 = frag_at(pa, 0, 8);
  v16h b0[4], b1[4];
  #pragma unroll
  for (int t = 0; t < 4; ++t) b0[t] = frag_at(pb[t], 0, 8);
  v16h a1 = frag_at(pa, 32, 8);
  #pragma unroll
  for (int t = 0; t < 4; ++t) b1[t] = frag_at(pb[t], 32, 8);

  // steady state: compute stage, then refill the SAME registers for kk+64 (no rotation movs)
  int kk = 0;
  for (; kk < K - 64; kk += 64) {
    #pragma unroll
    for (int t = 0; t < 4; ++t) acc[t] = wmma16(a0, b0[t], acc[t]);
    a0 = frag_at(pa, kk + 64, 8);
    #pragma unroll
    for (int t = 0; t < 4; ++t) b0[t] = frag_at(pb[t], kk + 64, 8);
    #pragma unroll
    for (int t = 0; t < 4; ++t) acc[t] = wmma16(a1, b1[t], acc[t]);
    a1 = frag_at(pa, kk + 96, 8);
    #pragma unroll
    for (int t = 0; t < 4; ++t) b1[t] = frag_at(pb[t], kk + 96, 8);
  }
  // epilogue: last two stages
  #pragma unroll
  for (int t = 0; t < 4; ++t) acc[t] = wmma16(a0, b0[t], acc[t]);
  #pragma unroll
  for (int t = 0; t < 4; ++t) acc[t] = wmma16(a1, b1[t], acc[t]);

  // store (C/D layout: elem r -> row m0 + (lane>>4)*8 + r, col n0 + lane&15)
  int rb = m0 + ((lane >> 4) << 3);
  int cb = n0 + (lane & 15);
  #pragma unroll
  for (int t = 0; t < 4; ++t) {
    int c = cb + 16 * t;
    float bv = bias ? bias[c] : 0.0f;
    #pragma unroll
    for (int r = 0; r < 8; ++r) D[(size_t)(rb + r) * ldd + c] = acc[t][r] + bv;
  }
}

// ---------------- 6) split qkv -> q16/k16 [b,h,kt,dh], v16t [b,h,dh,kt] ----------------
__global__ void __launch_bounds__(256) qkv_split_kernel(
    const float* __restrict__ qkv, _Float16* __restrict__ q16,
    _Float16* __restrict__ k16, _Float16* __restrict__ v16t) {
  int b = blockIdx.y, kk = blockIdx.x, tid = threadIdx.x;
  size_t row = (size_t)(b * Kc + kk);
  int c0 = tid * 4;                       // 256 threads * 4 = 1024 = C
  int hh = c0 >> 6, d = c0 & 63;
  size_t base = (size_t)(b * NH + hh);
  v4f qv = *(const v4f*)(qkv + row * 3 * Cc + c0);
  v4f kv = *(const v4f*)(qkv + row * 3 * Cc + Cc + c0);
  v4f vv = *(const v4f*)(qkv + row * 3 * Cc + 2 * Cc + c0);
  v4h qo, ko;
  #pragma unroll
  for (int i = 0; i < 4; ++i) { qo[i] = (_Float16)qv[i]; ko[i] = (_Float16)kv[i]; }
  *(v4h*)(q16 + (base * Kc + kk) * DHc + d) = qo;
  *(v4h*)(k16 + (base * Kc + kk) * DHc + d) = ko;
  #pragma unroll
  for (int i = 0; i < 4; ++i)
    v16t[(base * DHc + d + i) * Kc + kk] = (_Float16)vv[i];
}

// ---------------- 7) causal flash attention (wave = 16 q-rows), pipelined ----------------
__global__ void __launch_bounds__(256) attn_kernel(
    const _Float16* __restrict__ q16, const _Float16* __restrict__ k16,
    const _Float16* __restrict__ v16t, _Float16* __restrict__ attn16) {
  __shared__ __align__(16) _Float16 pst[8][16 * 40];   // per-wave P staging (16x32, stride 40)
  int wave = threadIdx.x >> 5, lane = threadIdx.x & 31;
  int b = blockIdx.z, hh = blockIdx.y;
  int q0 = blockIdx.x * 128 + wave * 16;
  size_t base = (size_t)(b * NH + hh);
  const _Float16* Q  = q16  + base * Kc * DHc;   // [Kc, DH]
  const _Float16* Km = k16  + base * Kc * DHc;   // [Kc, DH]
  const _Float16* Vt = v16t + base * DHc * Kc;   // [DH, Kc]
  _Float16* myp = &pst[wave][0];

  int colbase = lane & 15;
  int rbloc = (lane >> 4) << 3;

  // per-lane base pointers
  const _Float16* pQ = Q + (size_t)(q0 + colbase) * DHc + ((lane >> 4) << 4);
  const _Float16* pK = Km + (size_t)colbase * DHc + ((lane >> 4) << 4);
  const _Float16* pV = Vt + (size_t)colbase * Kc + ((lane >> 4) << 4);
  const _Float16* pP = myp + colbase * 40 + ((lane >> 4) << 4);

  v16h qa0 = frag_at(pQ, 0, 8);
  v16h qa1 = frag_at(pQ, 32, 8);

  v8f o[4] = {};
  float mrow[8], lrow[8];
  #pragma unroll
  for (int r = 0; r < 8; ++r) { mrow[r] = -1e30f; lrow[r] = 0.f; }

  int kend = q0 + 16;                 // exclusive causal bound for this tile

  // prologue: K fragments for first chunk: [tile*2 + kstep]
  v16h kf[4];
  kf[0] = frag_at(pK, 0, 8);
  kf[1] = frag_at(pK, 32, 8);
  kf[2] = frag_at(pK + 16 * DHc, 0, 8);
  kf[3] = frag_at(pK + 16 * DHc, 32, 8);

  for (int kb = 0; kb < kend; kb += 32) {
    // ---- S = Q @ K^T for 32 key columns (consumes kf) ----
    v8f sv[2] = {};
    sv[0] = wmma16(qa0, kf[0], sv[0]);
    sv[0] = wmma16(qa1, kf[1], sv[0]);
    sv[1] = wmma16(qa0, kf[2], sv[1]);
    sv[1] = wmma16(qa1, kf[3], sv[1]);
    // V fragments for this chunk (overlap with softmax VALU work)
    v16h vf[4];
    #pragma unroll
    for (int tn = 0; tn < 4; ++tn) vf[tn] = frag_at(pV + tn * 16 * Kc, kb, 8);
    // refill kf in place for the next chunk (kf already consumed above)
    bool more = (kb + 32) < kend;
    if (more) {
      const _Float16* pKn = pK + (kb + 32) * DHc;
      kf[0] = frag_at(pKn, 0, 8);
      kf[1] = frag_at(pKn, 32, 8);
      kf[2] = frag_at(pKn + 16 * DHc, 0, 8);
      kf[3] = frag_at(pKn + 16 * DHc, 32, 8);
    }
    // ---- online softmax update (per row, cross-lane over the 16 col-lanes) ----
    #pragma unroll
    for (int r = 0; r < 8; ++r) {
      int qrow = q0 + rbloc + r;
      #pragma unroll
      for (int t = 0; t < 2; ++t) {
        int col = kb + t * 16 + colbase;
        float s = sv[t][r] * 0.125f;             // 1/sqrt(64)
        if (col > qrow) s = -1e30f;
        sv[t][r] = s;
      }
      float mx = fmaxf(sv[0][r], sv[1][r]);
      #pragma unroll
      for (int off = 1; off < 16; off <<= 1) mx = fmaxf(mx, __shfl_xor(mx, off, 32));
      float mnew = fmaxf(mrow[r], mx);
      float alpha = __expf(mrow[r] - mnew);
      float p0 = __expf(sv[0][r] - mnew);
      float p1 = __expf(sv[1][r] - mnew);
      float ps = p0 + p1;
      #pragma unroll
      for (int off = 1; off < 16; off <<= 1) ps += __shfl_xor(ps, off, 32);
      lrow[r] = lrow[r] * alpha + ps;
      mrow[r] = mnew;
      #pragma unroll
      for (int t = 0; t < 4; ++t) o[t][r] *= alpha;
      sv[0][r] = p0; sv[1][r] = p1;
    }
    // ---- stage P (16x32) into LDS, re-load as A-fragment ----
    #pragma unroll
    for (int t = 0; t < 2; ++t)
      #pragma unroll
      for (int r = 0; r < 8; ++r)
        myp[(rbloc + r) * 40 + t * 16 + colbase] = (_Float16)sv[t][r];
    v16h pa = frag_at(pP, 0, 8);
    // ---- O += P @ V  (V^T stored [DH, Kc] -> TN B operand) ----
    #pragma unroll
    for (int tn = 0; tn < 4; ++tn) o[tn] = wmma16(pa, vf[tn], o[tn]);
  }

  // ---- normalize and write attn output as f16 into [B*Kc, C] ----
  #pragma unroll
  for (int r = 0; r < 8; ++r) {
    float inv = 1.0f / lrow[r];
    size_t row = (size_t)(b * Kc + q0 + rbloc + r);
    #pragma unroll
    for (int t = 0; t < 4; ++t)
      attn16[row * Cc + hh * 64 + t * 16 + colbase] = (_Float16)(o[t][r] * inv);
  }
}

// ---------------- 8) residual add + LN2 (float4) ----------------
__global__ void __launch_bounds__(256) resid_ln2_kernel(
    const float* __restrict__ ao, const float* __restrict__ g, const float* __restrict__ be,
    float* __restrict__ h, _Float16* __restrict__ mln) {
  __shared__ float red[8];
  int b = blockIdx.y, kk = blockIdx.x, tid = threadIdx.x;
  size_t row = (size_t)(b * Kc + kk);
  float* hr = h + row * Cc;
  const float* ar = ao + row * Cc;
  int c0 = tid * 4;
  v4f v = *(const v4f*)(hr + c0);
  v4f a = *(const v4f*)(ar + c0);
  #pragma unroll
  for (int i = 0; i < 4; ++i) v[i] += a[i];
  *(v4f*)(hr + c0) = v;
  float s = v[0] + v[1] + v[2] + v[3];
  float mean = block_sum256(s, red, tid) * (1.0f / Cc);
  float d = 0.f;
  #pragma unroll
  for (int i = 0; i < 4; ++i) { float t = v[i] - mean; d += t * t; }
  float var = block_sum256(d, red, tid) * (1.0f / Cc);
  float rstd = rsqrtf(var + 1e-5f);
  v4f gv = *(const v4f*)(g + c0);
  v4f bv = *(const v4f*)(be + c0);
  v4h o;
  #pragma unroll
  for (int i = 0; i < 4; ++i) o[i] = (_Float16)((v[i] - mean) * rstd * gv[i] + bv[i]);
  *(v4h*)(mln + row * Cc + c0) = o;
}

// ---------------- 9) exact GELU -> f16 (x4 vectorized) ----------------
__global__ void __launch_bounds__(256) gelu_kernel(
    const float* __restrict__ src, _Float16* __restrict__ dst, long n4) {
  long i = (long)blockIdx.x * 256 + threadIdx.x;
  long stride = (long)gridDim.x * 256;
  for (; i < n4; i += stride) {
    v4f v = *(const v4f*)(src + i * 4);
    v4h o;
    #pragma unroll
    for (int j = 0; j < 4; ++j)
      o[j] = (_Float16)(0.5f * v[j] * (1.0f + erff(v[j] * 0.70710678118654752f)));
    *(v4h*)(dst + i * 4) = o;
  }
}

// ---------------- 10) out = x (float4) ----------------
__global__ void __launch_bounds__(256) copy_kernel(
    const float* __restrict__ src, float* __restrict__ dst, long n4) {
  long i = (long)blockIdx.x * 256 + threadIdx.x;
  long stride = (long)gridDim.x * 256;
  for (; i < n4; i += stride)
    *(v4f*)(dst + i * 4) = *(const v4f*)(src + i * 4);
}

// ---------------- 11) weighted scatter-add (indices unique per batch), float4 ----------------
__global__ void __launch_bounds__(256) scatter_kernel(
    float* __restrict__ out, const int* __restrict__ idx, const float* __restrict__ wgt,
    const float* __restrict__ h, const float* __restrict__ mo) {
  int b = blockIdx.y, kk = blockIdx.x, tid = threadIdx.x;
  int tok = idx[b * Kc + kk];
  float w = wgt[b * Kc + kk];
  size_t row = (size_t)(b * Kc + kk);
  float* dst = out + ((size_t)b * Tc + tok) * Cc;
  int c0 = tid * 4;
  v4f hv = *(const v4f*)(h + row * Cc + c0);
  v4f mv = *(const v4f*)(mo + row * Cc + c0);
  v4f dv = *(v4f*)(dst + c0);
  #pragma unroll
  for (int i = 0; i < 4; ++i) dv[i] += (hv[i] + mv[i]) * w;
  *(v4f*)(dst + c0) = dv;
}

// ---------------- host-side launcher ----------------
extern "C" void kernel_launch(void* const* d_in, const int* in_sizes, int n_in,
                              void* d_out, int out_size, void* d_ws, size_t ws_size,
                              hipStream_t stream) {
  const float* x      = (const float*)d_in[0];
  const float* wr     = (const float*)d_in[1];
  const float* ln1w   = (const float*)d_in[2];
  const float* ln1b   = (const float*)d_in[3];
  const float* wqkv   = (const float*)d_in[4];
  const float* bqkv   = (const float*)d_in[5];
  const float* wo     = (const float*)d_in[6];
  const float* bo     = (const float*)d_in[7];
  const float* ln2w   = (const float*)d_in[8];
  const float* ln2b   = (const float*)d_in[9];
  const float* wfc    = (const float*)d_in[10];
  const float* bfc    = (const float*)d_in[11];
  const float* wproj  = (const float*)d_in[12];
  const float* bproj  = (const float*)d_in[13];
  float* out = (float*)d_out;

  char* ws = (char*)d_ws;
  size_t off = 0;
  auto take = [&](size_t bytes) -> char* {
    char* p = ws + off;
    off += (bytes + 255) & ~(size_t)255;
    return p;
  };

  int*      idxW   = (int*)  take(BK * sizeof(int));
  float*    wgtW   = (float*)take(BK * sizeof(float));
  float*    logits = (float*)take((size_t)Bc * Tc * sizeof(float));
  float*    hbuf   = (float*)take((size_t)BK * Cc * sizeof(float));       // residual stream (fp32)
  _Float16* wqkv16 = (_Float16*)take((size_t)3 * Cc * Cc * 2);
  _Float16* wo16   = (_Float16*)take((size_t)Cc * Cc * 2);
  _Float16* wfc16  = (_Float16*)take((size_t)DFFc * Cc * 2);
  _Float16* wpj16  = (_Float16*)take((size_t)Cc * DFFc * 2);
  _Float16* aln16  = (_Float16*)take((size_t)BK * Cc * 2);
  _Float16* q16    = (_Float16*)take((size_t)BK * Cc * 2);    // [b,h,kt,dh]
  _Float16* k16    = (_Float16*)take((size_t)BK * Cc * 2);    // [b,h,kt,dh]
  _Float16* v16t   = (_Float16*)take((size_t)BK * Cc * 2);    // [b,h,dh,kt]
  _Float16* attn16 = (_Float16*)take((size_t)BK * Cc * 2);    // [B*Kc, C]
  float*    big    = (float*)take((size_t)BK * DFFc * sizeof(float)); // qkv / ao / fc / mo (time-shared)
  _Float16* g16    = q16;      // alias: 4x16MB contiguous = BK*DFF f16 (q/k/v/attn dead by then)
  _Float16* mln16  = attn16;   // alias: attn dead after o-proj

  // 1) router logits
  router_logits_kernel<<<dim3(Bc * Tc / 8), 256, 0, stream>>>(x, wr, logits);
  // 2) top-k (sorted by position)
  topk_kernel<<<dim3(Bc), 1024, 0, stream>>>(logits, idxW, wgtW);
  // 3) gather + LN1
  gather_ln1_kernel<<<dim3(Kc, Bc), 256, 0, stream>>>(x, idxW, ln1w, ln1b, hbuf, aln16);
  // 4) weights to f16
  f32_to_f16_kernel<<<dim3(1024), 256, 0, stream>>>(wqkv,  wqkv16, (long)3 * Cc * Cc / 4);
  f32_to_f16_kernel<<<dim3(512),  256, 0, stream>>>(wo,    wo16,   (long)Cc * Cc / 4);
  f32_to_f16_kernel<<<dim3(1024), 256, 0, stream>>>(wfc,   wfc16,  (long)DFFc * Cc / 4);
  f32_to_f16_kernel<<<dim3(1024), 256, 0, stream>>>(wproj, wpj16,  (long)Cc * DFFc / 4);
  // 5) QKV GEMM: [8192,1024] x [3072,1024]^T + b_qkv
  wmma_gemm_tn<<<dim3(3 * Cc / 64, BK / 128), 256, 0, stream>>>(
      big, aln16, wqkv16, bqkv, (int)BK, 3 * Cc, Cc, Cc, Cc, 3 * Cc);
  // 6) split into per-head q/k/v (v transposed)
  qkv_split_kernel<<<dim3(Kc, Bc), 256, 0, stream>>>(big, q16, k16, v16t);
  // 7) causal attention
  attn_kernel<<<dim3(Kc / 128, NH, Bc), 256, 0, stream>>>(q16, k16, v16t, attn16);
  // 8) output projection: + b_o
  wmma_gemm_tn<<<dim3(Cc / 64, BK / 128), 256, 0, stream>>>(
      big, attn16, wo16, bo, (int)BK, Cc, Cc, Cc, Cc, Cc);
  // 9) residual + LN2
  resid_ln2_kernel<<<dim3(Kc, Bc), 256, 0, stream>>>(big, ln2w, ln2b, hbuf, mln16);
  // 10) FC GEMM: [8192,1024] x [4096,1024]^T + b_fc
  wmma_gemm_tn<<<dim3(DFFc / 64, BK / 128), 256, 0, stream>>>(
      big, mln16, wfc16, bfc, (int)BK, DFFc, Cc, Cc, Cc, DFFc);
  // 11) exact GELU -> f16
  gelu_kernel<<<dim3(8192), 256, 0, stream>>>(big, g16, BK * DFFc / 4);
  // 12) proj GEMM: [8192,4096] x [1024,4096]^T + b_proj
  wmma_gemm_tn<<<dim3(Cc / 64, BK / 128), 256, 0, stream>>>(
      big, g16, wpj16, bproj, (int)BK, Cc, DFFc, DFFc, DFFc, Cc);
  // 13) out = x
  copy_kernel<<<dim3(8192), 256, 0, stream>>>(x, out, (long)Bc * Tc * Cc / 4);
  // 14) out[b, idx, :] += (h + mlp_out) * weight
  scatter_kernel<<<dim3(Kc, Bc), 256, 0, stream>>>(out, idxW, wgtW, hbuf, big);

  (void)in_sizes; (void)n_in; (void)out_size; (void)ws_size;
}
